// ada_gcn_79963701117631
// MI455X (gfx1250) — matrix operand order repaired and checked by reading, not compiled, not run on hardware
//
#include <hip/hip_runtime.h>
#include <math.h>

// ---------------- types for WMMA ----------------
typedef __bf16 bf16;
typedef __attribute__((ext_vector_type(16))) __bf16 v16bf;
typedef __attribute__((ext_vector_type(8)))  __bf16 v8bf;
typedef __attribute__((ext_vector_type(8)))  float  v8f;

#define WMMA_BF16(a, b, c) \
  __builtin_amdgcn_wmma_f32_16x16x32_bf16(false, (a), false, (b), (short)0, (c), false, false)

#define Bq 32
#define Nn 512
#define Hh 4
#define Cc 32

static __device__ inline v16bf cat8(v8bf lo, v8bf hi) {
  return __builtin_shufflevector(lo, hi, 0,1,2,3,4,5,6,7,8,9,10,11,12,13,14,15);
}

// A-fragment (16x32 bf16): laneRow = row (lane&15) base; per-lane K runs {h8..h8+7, 16+h8..+7}
static __device__ inline v16bf fragA_from(const bf16* laneRow, int k0, int h8) {
  v8bf lo = *reinterpret_cast<const v8bf*>(laneRow + k0 + h8);
  v8bf hi = *reinterpret_cast<const v8bf*>(laneRow + k0 + 16 + h8);
  return cat8(lo, hi);
}
// B-fragment (32x16 bf16) from column-major storage: laneRow = column (lane&15) base;
// per-lane K runs {koffB..koffB+7, 16+koffB..+7}
static __device__ inline v16bf fragB_from(const bf16* laneRow, int k0, int koffB) {
  v8bf lo = *reinterpret_cast<const v8bf*>(laneRow + k0 + koffB);
  v8bf hi = *reinterpret_cast<const v8bf*>(laneRow + k0 + koffB + 16);
  return cat8(lo, hi);
}
// legacy row-major helpers (attention kernel)
static __device__ inline v16bf load_fragA(const bf16* base, int rowStride) {
  int lane = threadIdx.x & 31;
  return fragA_from(base + (size_t)(lane & 15) * rowStride, 0, (lane >> 4) * 8);
}
static __device__ inline v16bf load_fragB(const bf16* base, int rowStride) {
  int lane = threadIdx.x & 31;
  return fragB_from(base + (size_t)(lane & 15) * rowStride, 0, (lane >> 4) * 8);
}

// ---------------- K1: feature -> l,r (bf16) ----------------
__global__ void lr_kernel(const float* __restrict__ inp, const float* __restrict__ emb,
                          const float* __restrict__ ew, const float* __restrict__ ebias,
                          const float* __restrict__ wlw, const float* __restrict__ wlb,
                          const float* __restrict__ wrw, const float* __restrict__ wrb,
                          bf16* __restrict__ lbuf, bf16* __restrict__ rbuf) {
  int gid = blockIdx.x * blockDim.x + threadIdx.x;
  if (gid >= Bq * Nn) return;
  int b = gid / Nn, n = gid % Nn;
  float dot = ebias[0];
  for (int t = 0; t < 12; ++t) dot += inp[((size_t)b * 12 + t) * Nn + n] * ew[t];
  float feat[12];
  for (int j = 0; j < 12; ++j) feat[j] = dot + emb[n * 12 + j];
  for (int hc = 0; hc < 128; ++hc) {
    float lv = wlb[hc], rv = wrb[hc];
    for (int j = 0; j < 12; ++j) { lv += feat[j] * wlw[j * 128 + hc]; rv += feat[j] * wrw[j * 128 + hc]; }
    size_t oi = ((size_t)(b * Hh + (hc >> 5)) * Nn + n) * 32 + (hc & 31);
    lbuf[oi] = (bf16)lv;
    rbuf[oi] = (bf16)rv;
  }
}

// ---------------- K2: fused attention (WMMA) + top-k + softmax -> dynT (bf16) ----------------
__global__ __launch_bounds__(128) void attn_topk_kernel(const bf16* __restrict__ lbuf,
                                                        const bf16* __restrict__ rbuf,
                                                        bf16* __restrict__ dynT) {
  __shared__ float attS[16][Nn + 8];
  int blk = blockIdx.x;
  int nt = blk & 31;
  int h = (blk >> 5) & 3;
  int b = blk >> 7;
  int n0 = nt * 16;
  int wave = threadIdx.x >> 5;
  int lane = threadIdx.x & 31;
  const float scale = 0.17677669529663687f;  // 1/sqrt(32)

  const bf16* lrow = lbuf + ((size_t)(b * Hh + h) * Nn + n0) * 32;
  v16bf afrag = load_fragA(lrow, 32);
  for (int ct = wave; ct < 32; ct += 4) {
    int m0 = ct * 16;
    // B operand: Bs[k][col] = r[m0+col][k]  == r row-major directly
    const bf16* rrow = rbuf + ((size_t)(b * Hh + h) * Nn + m0) * 32;
    v16bf bfrag = load_fragB(rrow, 32);
    v8f acc = {0.f, 0.f, 0.f, 0.f, 0.f, 0.f, 0.f, 0.f};
    acc = WMMA_BF16(afrag, bfrag, acc);
    int n = lane & 15, r8 = (lane >> 4) * 8;
    for (int j = 0; j < 8; ++j) attS[j + r8][m0 + n] = acc[j] * scale;
  }
  __syncthreads();

  int kk = (h == 0) ? 10 : (h == 1) ? 20 : (h == 2) ? 40 : 500;
  for (int row = wave; row < 16; row += 4) {
    // exact k-th-largest via binary search on monotone float key
    unsigned lo = 0u, hi = 0xFFFFFFFFu;
    for (int it = 0; it < 32; ++it) {
      unsigned mid = (unsigned)(((unsigned long long)lo + (unsigned long long)hi + 1ull) >> 1);
      int cnt = 0;
      for (int e = 0; e < 16; ++e) {
        unsigned u = __float_as_uint(attS[row][lane + 32 * e]);
        unsigned key = (u & 0x80000000u) ? ~u : (u ^ 0x80000000u);
        cnt += (key >= mid) ? 1 : 0;
      }
      for (int off = 16; off; off >>= 1) cnt += __shfl_xor(cnt, off, 32);
      if (cnt >= kk) lo = mid; else hi = mid - 1;
    }
    unsigned thr = lo;
    float mx = -1e30f;
    for (int e = 0; e < 16; ++e) {
      float v = attS[row][lane + 32 * e];
      unsigned u = __float_as_uint(v);
      unsigned key = (u & 0x80000000u) ? ~u : (u ^ 0x80000000u);
      if (key >= thr) mx = fmaxf(mx, v);
    }
    for (int off = 16; off; off >>= 1) mx = fmaxf(mx, __shfl_xor(mx, off, 32));
    float sm = 0.f;
    for (int e = 0; e < 16; ++e) {
      float v = attS[row][lane + 32 * e];
      unsigned u = __float_as_uint(v);
      unsigned key = (u & 0x80000000u) ? ~u : (u ^ 0x80000000u);
      if (key >= thr) sm += __expf(v - mx);
    }
    for (int off = 16; off; off >>= 1) sm += __shfl_xor(sm, off, 32);
    float inv = 1.f / sm;
    // write transposed: dynT[b,h,m,v] = softmax(att)[v,m]
    bf16* dcol = dynT + ((size_t)(b * Hh + h) * Nn) * Nn + (n0 + row);
    for (int e = 0; e < 16; ++e) {
      int m = lane + 32 * e;
      float v = attS[row][m];
      unsigned u = __float_as_uint(v);
      unsigned key = (u & 0x80000000u) ? ~u : (u ^ 0x80000000u);
      float o = (key >= thr) ? __expf(v - mx) * inv : 0.f;
      dcol[(size_t)m * Nn] = (bf16)o;
    }
  }
}

// ---------------- K3: supports -> supT (transposed, slots 0,2) + supB (plain bf16) --------
__global__ void supcvt_kernel(const float* __restrict__ sup, bf16* __restrict__ supT,
                              bf16* __restrict__ supB) {
  long gid = blockIdx.x * (long)blockDim.x + threadIdx.x;
  if (gid >= 2L * Nn * Nn) return;
  int s = (int)(gid / ((long)Nn * Nn));
  long rem = gid % ((long)Nn * Nn);
  int v = (int)(rem / Nn), w = (int)(rem % Nn);
  float val = sup[(size_t)s * Nn * Nn + (size_t)v * Nn + w];
  supT[(size_t)(s * 2) * Nn * Nn + (size_t)w * Nn + v] = (bf16)val;
  supB[(size_t)s * Nn * Nn + (size_t)v * Nn + w] = (bf16)val;
}

// ---------------- K4: batched bf16 WMMA GEMM, fragments straight from global --------------
// C[z](M x Ncols) = A[z](M x K, row-major) * B[z](K x Ncols, stored COLUMN-major: BT[col][k])
// wave tile 32x64 (8 WMMA / K-step), block tile 128x64 (4 waves stacked in M)
__global__ __launch_bounds__(128) void gemm_wmma_kernel(
    const bf16* __restrict__ A, const bf16* __restrict__ BT, bf16* __restrict__ C,
    int M, int Ncols, int K,
    long long aStride, int aMod, long long bStride, int bDiv,
    long long cStride, long long cBase) {
  int z = blockIdx.z;
  const bf16* Ap = A + (size_t)(aMod ? (z % aMod) : z) * aStride;
  const bf16* Bp = BT + (size_t)(z / bDiv) * bStride;
  bf16* Cp = C + cBase + (size_t)z * cStride;
  int wave = threadIdx.x >> 5, lane = threadIdx.x & 31;
  int m0 = blockIdx.y * 128 + wave * 32;
  int c0 = blockIdx.x * 64;
  int mlane = lane & 15;
  int h8 = (lane >> 4) * 8;
  int koffB = (lane >> 4) * 8;

  const bf16* arow0 = Ap + (size_t)(m0 + mlane) * K;
  const bf16* arow1 = arow0 + (size_t)16 * K;
  const bf16* brow[4];
#pragma unroll
  for (int tj = 0; tj < 4; ++tj) {
    int j = c0 + tj * 16 + mlane;
    if (j > Ncols - 1) j = Ncols - 1;  // clamp (stores are guarded)
    brow[tj] = Bp + (size_t)j * K;
  }
  v8f z8 = {0.f, 0.f, 0.f, 0.f, 0.f, 0.f, 0.f, 0.f};
  v8f acc[2][4];
#pragma unroll
  for (int ti = 0; ti < 2; ++ti)
#pragma unroll
    for (int tj = 0; tj < 4; ++tj) acc[ti][tj] = z8;

  for (int k0 = 0; k0 < K; k0 += 32) {
    if (k0 + 32 < K) {
      __builtin_prefetch(arow0 + k0 + 32, 0, 1);
      __builtin_prefetch(brow[0] + k0 + 32, 0, 1);
    }
    v16bf fa0 = fragA_from(arow0, k0, h8);
    v16bf fa1 = fragA_from(arow1, k0, h8);
#pragma unroll
    for (int tj = 0; tj < 4; ++tj) {
      v16bf fb = fragB_from(brow[tj], k0, koffB);
      acc[0][tj] = WMMA_BF16(fa0, fb, acc[0][tj]);
      acc[1][tj] = WMMA_BF16(fa1, fb, acc[1][tj]);
    }
  }
  int n = lane & 15, r8 = (lane >> 4) * 8;
#pragma unroll
  for (int ti = 0; ti < 2; ++ti)
#pragma unroll
    for (int tj = 0; tj < 4; ++tj) {
      int rbase = m0 + ti * 16;
      int cidx = c0 + tj * 16 + n;
      if (cidx < Ncols) {
#pragma unroll
        for (int j = 0; j < 8; ++j)
          Cp[(size_t)(rbase + j + r8) * Ncols + cidx] = (bf16)acc[ti][tj][j];
      }
    }
}

// ---------------- K5: start conv (pad-left 1, 1x1) ----------------
__global__ void start_kernel(const float* __restrict__ inp, const float* __restrict__ sw,
                             const float* __restrict__ sb, float* __restrict__ x) {
  int gid = blockIdx.x * blockDim.x + threadIdx.x;
  if (gid >= Bq * Nn * 13) return;
  int t = gid % 13;
  int bn = gid / 13;
  int n = bn % Nn, b = bn / Nn;
  float v = (t == 0) ? 0.f : inp[((size_t)b * 12 + (t - 1)) * Nn + n];
  for (int c = 0; c < Cc; ++c)
    x[(((size_t)b * Cc + c) * Nn + n) * 13 + t] = sw[c] * v + sb[c];
}

// ---------------- K6: dilated conv + tanh*sigmoid gating ----------------
__global__ __launch_bounds__(128) void dconv_gate_kernel(const float* __restrict__ x,
                                                         const float* __restrict__ fw, const float* __restrict__ fb,
                                                         const float* __restrict__ gw, const float* __restrict__ gb,
                                                         float* __restrict__ out, int Tprev, int d) {
  __shared__ float wF[2048], wG[2048], bF[32], bG[32];
  int T = Tprev - d;
  for (int i = threadIdx.x; i < 2048; i += 128) { wF[i] = fw[i]; wG[i] = gw[i]; }
  if (threadIdx.x < 32) { bF[threadIdx.x] = fb[threadIdx.x]; bG[threadIdx.x] = gb[threadIdx.x]; }
  __syncthreads();
  long gid = blockIdx.x * 128L + threadIdx.x;
  long tot = (long)Bq * Nn * T;
  if (gid >= tot) return;
  int t = (int)(gid % T);
  long bn = gid / T;
  int n = (int)(bn % Nn), b = (int)(bn / Nn);
  float x0[32], x1[32];
  for (int c = 0; c < 32; ++c) {
    const float* p = x + (((size_t)b * Cc + c) * Nn + n) * Tprev;
    x0[c] = p[t];
    x1[c] = p[t + d];
  }
  for (int o = 0; o < 32; ++o) {
    float f = bF[o], g = bG[o];
    for (int c = 0; c < 32; ++c) {
      f += wF[(o * 32 + c) * 2] * x0[c] + wF[(o * 32 + c) * 2 + 1] * x1[c];
      g += wG[(o * 32 + c) * 2] * x0[c] + wG[(o * 32 + c) * 2 + 1] * x1[c];
    }
    out[(((size_t)b * Cc + o) * Nn + n) * T + t] = tanhf(f) * (1.f / (1.f + __expf(-g)));
  }
}

// ---------------- K7: skip conv, only last timestep survives to the head ----------------
__global__ void skip_kernel(const float* __restrict__ gated, const float* __restrict__ skw,
                            const float* __restrict__ skb, float* __restrict__ skipAcc,
                            int T, int first) {
  long gid = blockIdx.x * (long)blockDim.x + threadIdx.x;
  if (gid >= (long)Bq * Nn * 256) return;
  int oc = (int)(gid % 256);
  long bn = gid / 256;
  int n = (int)(bn % Nn), b = (int)(bn / Nn);
  float acc = skb[oc];
  for (int c = 0; c < 32; ++c)
    acc += skw[oc * 32 + c] * gated[(((size_t)b * Cc + c) * Nn + n) * T + (T - 1)];
  size_t oi = ((size_t)b * 256 + oc) * Nn + n;
  skipAcc[oi] = first ? acc : (skipAcc[oi] + acc);
}

// ---------------- K8: x (b,c,v,t) f32 -> xbT (b, j=c*T+t, v) bf16 (column-major B) -------
__global__ void prep_kernel(const float* __restrict__ x, bf16* __restrict__ xbT, int T) {
  int NC = Cc * T;
  long gid = blockIdx.x * (long)blockDim.x + threadIdx.x;
  long tot = (long)Bq * NC * Nn;
  if (gid >= tot) return;
  int v = (int)(gid % Nn);
  long bj = gid / Nn;
  int j = (int)(bj % NC);
  int b = (int)(bj / NC);
  int c = j / T, t = j % T;
  xbT[gid] = (bf16)x[(((size_t)b * Cc + c) * Nn + v) * T + t];
}

// ---------------- K9: gcn combine (async-LDS weight staging) ----------------
__global__ __launch_bounds__(128) void combine_kernel(const float* __restrict__ xin,
                                                      const bf16* __restrict__ agg,
                                                      const float* __restrict__ W,
                                                      const float* __restrict__ bias,
                                                      float* __restrict__ y, int T) {
  __shared__ __align__(16) float wS[32 * 160];
  __shared__ float xs[32 * 12];
  __shared__ float as[4 * 32 * 12];
  __shared__ float bS[32];
  int NC = Cc * T;
  int bw = blockIdx.x;
  int n = bw % Nn, b = bw / Nn;

  // async stage 20KB of weights: GLOBAL_LOAD_ASYNC_TO_LDS_B128, 16B per lane-op
  {
    unsigned ldsbase = (unsigned)(size_t)(void*)&wS[0];
    for (int i = threadIdx.x; i < 1280; i += 128) {
      unsigned dst = ldsbase + (unsigned)i * 16u;
      const float* src = W + (size_t)i * 4;
      asm volatile("global_load_async_to_lds_b128 %0, %1, off"
                   :: "v"(dst), "v"(src) : "memory");
    }
  }
  if (threadIdx.x < 32) bS[threadIdx.x] = bias[threadIdx.x];
  for (int i = threadIdx.x; i < NC; i += 128) {
    int c = i / T, t = i % T;
    xs[i] = xin[(((size_t)b * Cc + c) * Nn + n) * T + t];
  }
  for (int i = threadIdx.x; i < 4 * NC; i += 128) {
    int s = i / NC, j = i % NC;
    as[i] = (float)agg[((size_t)(b * 4 + s) * Nn + n) * NC + j];
  }
  asm volatile("s_wait_asynccnt 0x0" ::: "memory");
  __syncthreads();
  for (int idx = threadIdx.x; idx < NC; idx += 128) {
    int o = idx / T, t = idx % T;
    float acc = bS[o];
    const float* wr = &wS[o * 160];
    for (int c = 0; c < 32; ++c) acc += wr[c] * xs[c * T + t];
    for (int s = 0; s < 4; ++s) {
      const float* ap = &as[s * NC];
      const float* w2 = &wr[32 + s * 32];
      for (int c = 0; c < 32; ++c) acc += w2[c] * ap[c * T + t];
    }
    y[(((size_t)b * Cc + o) * Nn + n) * T + t] = acc;
  }
}

// ---------------- K10: residual + "bn" scale ----------------
__global__ void post_kernel(const float* __restrict__ y, const float* __restrict__ res,
                            const float* __restrict__ g, const float* __restrict__ bb,
                            float* __restrict__ xnext, int T, int Tprev, int d) {
  long gid = blockIdx.x * (long)blockDim.x + threadIdx.x;
  long tot = (long)Bq * Cc * Nn * T;
  if (gid >= tot) return;
  int t = (int)(gid % T);
  long r = gid / T;
  int n = (int)(r % Nn);
  long r2 = r / Nn;
  int c = (int)(r2 % Cc);
  int b = (int)(r2 / Cc);
  float v = y[gid] + res[(((size_t)b * Cc + c) * Nn + n) * Tprev + t + d];
  xnext[gid] = v * 0.9999950000374997f * g[c] + bb[c];  // 1/sqrt(1+1e-5)
}

// ---------------- K11: end head: relu -> 512 -> relu -> 12 ----------------
__global__ __launch_bounds__(128) void head_kernel(const float* __restrict__ skipAcc,
                                                   const float* __restrict__ e1w, const float* __restrict__ e1b,
                                                   const float* __restrict__ e2w, const float* __restrict__ e2b,
                                                   float* __restrict__ out) {
  __shared__ float sk[256];
  __shared__ float hS[512];
  int bw = blockIdx.x;
  int n = bw % Nn, b = bw / Nn;
  for (int i = threadIdx.x; i < 256; i += 128)
    sk[i] = fmaxf(0.f, skipAcc[((size_t)b * 256 + i) * Nn + n]);
  __syncthreads();
  for (int oh = threadIdx.x; oh < 512; oh += 128) {
    float acc = e1b[oh];
    const float* w = &e1w[(size_t)oh * 256];
    for (int sc = 0; sc < 256; ++sc) acc += w[sc] * sk[sc];
    hS[oh] = fmaxf(0.f, acc);
  }
  __syncthreads();
  if (threadIdx.x < 12) {
    int od = threadIdx.x;
    float acc = e2b[od];
    const float* w = &e2w[(size_t)od * 512];
    for (int oh = 0; oh < 512; ++oh) acc += w[oh] * hS[oh];
    out[((size_t)b * 12 + od) * Nn + n] = acc;
  }
}

// ---------------- host orchestration ----------------
extern "C" void kernel_launch(void* const* d_in, const int* in_sizes, int n_in,
                              void* d_out, int out_size, void* d_ws, size_t ws_size,
                              hipStream_t stream) {
  const float* input    = (const float*)d_in[0];
  const float* supports = (const float*)d_in[1];
  const float* emb      = (const float*)d_in[2];
  const float* embds_w  = (const float*)d_in[3];
  const float* embds_b  = (const float*)d_in[4];
  const float* wl_w     = (const float*)d_in[5];
  const float* wl_b     = (const float*)d_in[6];
  const float* wr_w     = (const float*)d_in[7];
  const float* wr_b     = (const float*)d_in[8];
  const float* start_w  = (const float*)d_in[9];
  const float* start_b  = (const float*)d_in[10];
  const float* filter_w = (const float*)d_in[11];
  const float* filter_b = (const float*)d_in[12];
  const float* gate_w   = (const float*)d_in[13];
  const float* gate_b   = (const float*)d_in[14];
  const float* skip_w   = (const float*)d_in[15];
  const float* skip_b   = (const float*)d_in[16];
  const float* bn_g     = (const float*)d_in[17];
  const float* bn_b     = (const float*)d_in[18];
  const float* gconv_w  = (const float*)d_in[19];
  const float* gconv_b  = (const float*)d_in[20];
  const float* dyn_w    = (const float*)d_in[21];
  const float* dyn_b    = (const float*)d_in[22];
  const float* end1_w   = (const float*)d_in[23];
  const float* end1_b   = (const float*)d_in[24];
  const float* end2_w   = (const float*)d_in[25];
  const float* end2_b   = (const float*)d_in[26];
  (void)in_sizes; (void)n_in; (void)out_size; (void)ws_size;

  char* ws = (char*)d_ws;
  size_t off = 0;
  auto alloc = [&](size_t bytes) -> char* {
    char* p = ws + off;
    off += (bytes + 255) & ~(size_t)255;
    return p;
  };
  bf16* lbuf = (bf16*)alloc(sizeof(bf16) * (size_t)Bq * Hh * Nn * 32);
  bf16* rbuf = (bf16*)alloc(sizeof(bf16) * (size_t)Bq * Hh * Nn * 32);
  bf16* dynT = (bf16*)alloc(sizeof(bf16) * (size_t)Bq * Hh * Nn * Nn);
  bf16* supT = (bf16*)alloc(sizeof(bf16) * (size_t)4 * Nn * Nn);
  bf16* supB = (bf16*)alloc(sizeof(bf16) * (size_t)2 * Nn * Nn);
  bf16* xbT  = (bf16*)alloc(sizeof(bf16) * (size_t)Bq * Cc * 12 * Nn);
  bf16* agg  = (bf16*)alloc(sizeof(bf16) * (size_t)Bq * 4 * Nn * Cc * 12);
  float* skipAcc = (float*)alloc(sizeof(float) * (size_t)Bq * 256 * Nn);
  float* bufA = (float*)alloc(sizeof(float) * (size_t)Bq * Cc * Nn * 13);
  float* bufB = (float*)alloc(sizeof(float) * (size_t)Bq * Cc * Nn * 13);
  float* bufC = (float*)alloc(sizeof(float) * (size_t)Bq * Cc * Nn * 13);

  // attention path
  lr_kernel<<<(Bq * Nn + 127) / 128, 128, 0, stream>>>(input, emb, embds_w, embds_b,
                                                       wl_w, wl_b, wr_w, wr_b, lbuf, rbuf);
  attn_topk_kernel<<<Bq * Hh * (Nn / 16), 128, 0, stream>>>(lbuf, rbuf, dynT);

  // static supports: convert, then square via (S*S)^T = S^T * S^T (B operand = plain S)
  supcvt_kernel<<<(2 * Nn * Nn + 255) / 256, 256, 0, stream>>>(supports, supT, supB);
  {
    dim3 g(Nn / 64, Nn / 128, 2);
    gemm_wmma_kernel<<<g, 128, 0, stream>>>(supT, supB, supT, Nn, Nn, Nn,
                                            (long long)2 * Nn * Nn, 0,
                                            (long long)Nn * Nn, 1,
                                            (long long)2 * Nn * Nn, (long long)Nn * Nn);
  }

  start_kernel<<<(Bq * Nn * 13 + 127) / 128, 128, 0, stream>>>(input, start_w, start_b, bufA);

  float* xcur = bufA;
  float* t1 = bufB;
  float* t2 = bufC;
  int Tprev = 13;
  const int dil[8] = {1, 2, 1, 2, 1, 2, 1, 2};
  for (int i = 0; i < 8; ++i) {
    int d = dil[i];
    int T = Tprev - d;
    int NC = Cc * T;
    dconv_gate_kernel<<<(int)(((long)Bq * Nn * T + 127) / 128), 128, 0, stream>>>(
        xcur, filter_w + (size_t)i * Cc * Cc * 2, filter_b + i * Cc,
        gate_w + (size_t)i * Cc * Cc * 2, gate_b + i * Cc, t1, Tprev, d);
    skip_kernel<<<(int)(((long)Bq * Nn * 256 + 255) / 256), 256, 0, stream>>>(
        t1, skip_w + (size_t)i * 256 * Cc, skip_b + i * 256, skipAcc, T, i == 0 ? 1 : 0);

    // dynamic-graph GCN
    prep_kernel<<<(int)(((long)Bq * NC * Nn + 255) / 256), 256, 0, stream>>>(t1, xbT, T);
    {
      dim3 g((NC + 63) / 64, Nn / 128, Bq * 4);
      gemm_wmma_kernel<<<g, 128, 0, stream>>>(dynT, xbT, agg, Nn, NC, Nn,
                                              (long long)Nn * Nn, 0,
                                              (long long)NC * Nn, 4,
                                              (long long)Nn * NC, 0);
    }
    combine_kernel<<<Bq * Nn, 128, 0, stream>>>(t1, agg, dyn_w + (size_t)i * Cc * 160,
                                                dyn_b + i * Cc, t2, T);

    // static-graph GCN
    prep_kernel<<<(int)(((long)Bq * NC * Nn + 255) / 256), 256, 0, stream>>>(t2, xbT, T);
    {
      dim3 g((NC + 63) / 64, Nn / 128, Bq * 4);
      gemm_wmma_kernel<<<g, 128, 0, stream>>>(supT, xbT, agg, Nn, NC, Nn,
                                              (long long)Nn * Nn, 4,
                                              (long long)NC * Nn, 4,
                                              (long long)Nn * NC, 0);
    }
    combine_kernel<<<Bq * Nn, 128, 0, stream>>>(t2, agg, gconv_w + (size_t)i * Cc * 160,
                                                gconv_b + i * Cc, t1, T);

    post_kernel<<<(int)(((long)Bq * Cc * Nn * T + 255) / 256), 256, 0, stream>>>(
        t1, xcur, bn_g + i * Cc, bn_b + i * Cc, t2, T, Tprev, d);

    float* newx = t2;
    t2 = xcur;
    xcur = newx;
    Tprev = T;
  }

  head_kernel<<<Bq * Nn, 128, 0, stream>>>(skipAcc, end1_w, end1_b, end2_w, end2_b, (float*)d_out);
}
